// GNNModel_25829933318392
// MI455X (gfx1250) — matrix operand order repaired and checked by reading, not compiled, not run on hardware
//
#include <hip/hip_runtime.h>
#include <math.h>

// ---------------------------------------------------------------------------
// GATv2 x2 + MLP head for MI455X (gfx1250, wave32, WMMA 16x16x32 f16)
// All WMMA operands pre-packed to f16 in lane-native layouts so the GEMM
// inner loop is pure b128 loads + v_wmma (no per-element bounds branches).
// ---------------------------------------------------------------------------

typedef __attribute__((ext_vector_type(16))) _Float16 v16h;
typedef __attribute__((ext_vector_type(8)))  _Float16 v8h;
typedef __attribute__((ext_vector_type(8)))  float    v8f;

static constexpr int NODES = 76000;
static constexpr int EDGES = 1216000;
static constexpr int H     = 5;
static constexpr int DH    = 16;
static constexpr int HD    = 80;     // H*DH
static constexpr int BROWS = 16000;  // (NODES/38)*8 rows for the head MLP

__device__ __forceinline__ float selu_f(float x) {
    const float sc = 1.0507009873554805f;
    const float sa = 1.7580993408473766f;   // scale*alpha
    return x > 0.f ? sc * x : sa * (expf(x) - 1.f);
}

__device__ __forceinline__ void atomic_max_f32(float* addr, float val) {
    unsigned int* ua = (unsigned int*)addr;
    unsigned int old = __float_as_uint(*addr);
    while (__uint_as_float(old) < val) {
        unsigned int assumed = old;
        old = atomicCAS(ua, assumed, __float_as_uint(val));
        if (old == assumed) break;
    }
}

// ---------------------------------------------------------------------------
// WMMA GEMM with pre-packed f16 operands.
//   A:  [M, lda] f16 row-major, lda multiple of 32, K zero-padded.
//   Bp: [KT][Ncpad][32] f16: Bp[(kt*Ncpad+n)*32 + j] = B[kt*32+j][n] (0-padded)
//   C = act(A @ B + bias): float (OUTH=0) or f16 (OUTH=1)
// One wave per 16x16 output tile. M is a multiple of 16 for every call site,
// so no m/k bounds checks; only the n<Nc store predicate remains.
// Lane layouts per CDNA5 ISA 7.12.2 (wave32).
// ---------------------------------------------------------------------------
template<int KT, int ACT, int OUTH>
__global__ __launch_bounds__(32)
void wmma_gemm_h(const _Float16* __restrict__ A, int lda,
                 const _Float16* __restrict__ Bp, int Ncpad,
                 const float* __restrict__ bias,
                 float* __restrict__ Cf, _Float16* __restrict__ Ch,
                 int ldc, int Nc)
{
    const int lane = threadIdx.x & 31;
    const int row0 = blockIdx.x << 4;
    const int n0   = blockIdx.y << 4;

    const int am  = row0 + (lane & 15);
    const int akb = (lane >> 4) << 3;   // 0 or 8
    const int bn  = n0 + (lane & 15);   // < Ncpad by construction
    const int bkb = (lane >> 4) << 3;   // 0 or 8

    const _Float16* __restrict__ Ap = A + (size_t)am * lda + akb;
    const _Float16* __restrict__ Bq = Bp + (size_t)bn * 32 + bkb;

    v8f acc = {0.f, 0.f, 0.f, 0.f, 0.f, 0.f, 0.f, 0.f};

#pragma unroll
    for (int kt = 0; kt < KT; ++kt) {
        const v8h alo = *(const v8h*)(Ap + kt * 32);
        const v8h ahi = *(const v8h*)(Ap + kt * 32 + 16);
        const v8h blo = *(const v8h*)(Bq + (size_t)kt * Ncpad * 32);
        const v8h bhi = *(const v8h*)(Bq + (size_t)kt * Ncpad * 32 + 16);
        const v16h a = __builtin_shufflevector(alo, ahi,
            0, 1, 2, 3, 4, 5, 6, 7, 8, 9, 10, 11, 12, 13, 14, 15);
        const v16h b = __builtin_shufflevector(blo, bhi,
            0, 1, 2, 3, 4, 5, 6, 7, 8, 9, 10, 11, 12, 13, 14, 15);
        acc = __builtin_amdgcn_wmma_f32_16x16x32_f16(
            false, a, false, b, (short)0, acc, false, false);
    }

    const bool  nok = bn < Nc;
    const float bv  = bias[nok ? bn : 0];
    const int   mb  = row0 + ((lane >> 4) << 3);
#pragma unroll
    for (int r = 0; r < 8; ++r) {
        if (nok) {
            float v = acc[r] + bv;
            if (ACT == 1) v = selu_f(v);
            if (OUTH) Ch[(size_t)(mb + r) * ldc + bn] = (_Float16)v;
            else      Cf[(size_t)(mb + r) * ldc + bn] = v;
        }
    }
}

// ---------------------------------------------------------------------------
// Operand packing kernels
// ---------------------------------------------------------------------------

// Weights [K, Nc] f32 -> Bp[(kt*Ncpad+n)*32 + j] f16 (zero-padded)
__global__ void pack_b(const float* __restrict__ B, _Float16* __restrict__ Bp,
                       int K, int Nc, int KT, int Ncpad)
{
    const int idx = blockIdx.x * blockDim.x + threadIdx.x;
    if (idx >= KT * Ncpad * 32) return;
    const int j   = idx & 31;
    const int rem = idx >> 5;
    const int n   = rem % Ncpad;
    const int kt  = rem / Ncpad;
    const int k   = kt * 32 + j;
    Bp[idx] = (_Float16)((k < K && n < Nc) ? B[(size_t)k * Nc + n] : 0.f);
}

// x [N,8] f32 -> xh [N,32] f16 (K padded 8->32)
__global__ void pack_x(const float* __restrict__ x, _Float16* __restrict__ xh)
{
    const int idx = blockIdx.x * blockDim.x + threadIdx.x;
    if (idx >= NODES * 32) return;
    const int n = idx >> 5;
    const int k = idx & 31;
    xh[idx] = (_Float16)((k < 8) ? x[(size_t)n * 8 + k] : 0.f);
}

// Transposed-reshape gather for fc1:
//   yt[r*64 + j] = y[(b*38 + j)*8 + c], r = b*8+c, j<38 (else 0)
__global__ void pack_yt(const float* __restrict__ y, _Float16* __restrict__ yt)
{
    const int idx = blockIdx.x * blockDim.x + threadIdx.x;
    if (idx >= BROWS * 64) return;
    const int r = idx >> 6;
    const int j = idx & 63;
    const int b = r >> 3;
    const int c = r & 7;
    yt[idx] = (_Float16)((j < 38) ? y[((size_t)b * 38 + j) * 8 + c] : 0.f);
}

// ---------------------------------------------------------------------------
// Edge pass 1: e[eg,h] = sum_k leaky_relu(xl[src]+xr[dst]) * att[h,k];
//              emax[dst,h] = max(...). One thread per (edge, head).
// ---------------------------------------------------------------------------
__device__ __forceinline__ float lrelu_dot4(float4 l, float4 r, float4 w) {
    float m0 = l.x + r.x; m0 = m0 > 0.f ? m0 : 0.2f * m0;
    float m1 = l.y + r.y; m1 = m1 > 0.f ? m1 : 0.2f * m1;
    float m2 = l.z + r.z; m2 = m2 > 0.f ? m2 : 0.2f * m2;
    float m3 = l.w + r.w; m3 = m3 > 0.f ? m3 : 0.2f * m3;
    return m0 * w.x + m1 * w.y + m2 * w.z + m3 * w.w;
}

__global__ void edge_scores(const float* __restrict__ xl,
                            const float* __restrict__ xr,
                            const int*   __restrict__ src,
                            const int*   __restrict__ dst,
                            const float* __restrict__ att,  // [H, DH]
                            float* __restrict__ esc,        // [E, H]
                            float* __restrict__ emax)       // [N, H]
{
    const int idx = blockIdx.x * blockDim.x + threadIdx.x;
    if (idx >= EDGES * H) return;
    const int eg = idx / H;
    const int h  = idx - eg * H;
    const int s  = src[eg];
    const int d  = dst[eg];

    const float4* pl = (const float4*)(xl  + (size_t)s * HD + h * DH);
    const float4* pr = (const float4*)(xr  + (size_t)d * HD + h * DH);
    const float4* pa = (const float4*)(att + h * DH);

    float acc = lrelu_dot4(pl[0], pr[0], pa[0])
              + lrelu_dot4(pl[1], pr[1], pa[1])
              + lrelu_dot4(pl[2], pr[2], pa[2])
              + lrelu_dot4(pl[3], pr[3], pa[3]);
    esc[idx] = acc;
    atomic_max_f32(emax + (size_t)d * H + h, acc);
}

// ---------------------------------------------------------------------------
// Edge pass 2: a = exp(e - emax[dst]); denom[dst,h] += a;
//              aggr[dst, h*16+k] += xl[src, h*16+k] * a
// ---------------------------------------------------------------------------
__global__ void edge_aggregate(const float* __restrict__ xl,
                               const int*   __restrict__ src,
                               const int*   __restrict__ dst,
                               const float* __restrict__ esc,
                               const float* __restrict__ emax,
                               float* __restrict__ denom,   // [N, H]
                               float* __restrict__ aggr)    // [N, HD]
{
    const int idx = blockIdx.x * blockDim.x + threadIdx.x;
    if (idx >= EDGES * H) return;
    const int eg = idx / H;
    const int h  = idx - eg * H;
    const int s  = src[eg];
    const int d  = dst[eg];

    const float a = expf(esc[idx] - emax[(size_t)d * H + h]);
    atomicAdd(denom + (size_t)d * H + h, a);

    const float4* pl = (const float4*)(xl + (size_t)s * HD + h * DH);
    float* po = aggr + (size_t)d * HD + h * DH;
    const float4 v0 = pl[0], v1 = pl[1], v2 = pl[2], v3 = pl[3];
    atomicAdd(po +  0, v0.x * a); atomicAdd(po +  1, v0.y * a);
    atomicAdd(po +  2, v0.z * a); atomicAdd(po +  3, v0.w * a);
    atomicAdd(po +  4, v1.x * a); atomicAdd(po +  5, v1.y * a);
    atomicAdd(po +  6, v1.z * a); atomicAdd(po +  7, v1.w * a);
    atomicAdd(po +  8, v2.x * a); atomicAdd(po +  9, v2.y * a);
    atomicAdd(po + 10, v2.z * a); atomicAdd(po + 11, v2.w * a);
    atomicAdd(po + 12, v3.x * a); atomicAdd(po + 13, v3.y * a);
    atomicAdd(po + 14, v3.z * a); atomicAdd(po + 15, v3.w * a);
}

// ---------------------------------------------------------------------------
// Node finalize: v = selu(aggr/(denom+1e-16) + bias), written as f16 into the
// concat buffer (and optionally the padded layer-2 GEMM input buffer).
// ---------------------------------------------------------------------------
__global__ void node_finalize(const float* __restrict__ aggr,
                              const float* __restrict__ denom,
                              const float* __restrict__ bias,   // [HD]
                              _Float16* __restrict__ out1, int ld1,
                              _Float16* __restrict__ out2, int ld2)
{
    const int idx = blockIdx.x * blockDim.x + threadIdx.x;
    if (idx >= NODES * HD) return;
    const int n = idx / HD;
    const int c = idx - n * HD;
    const int h = c >> 4;
    const float v = selu_f(aggr[idx] / (denom[(size_t)n * H + h] + 1e-16f) + bias[c]);
    out1[(size_t)n * ld1 + c] = (_Float16)v;
    if (out2) out2[(size_t)n * ld2 + c] = (_Float16)v;
}

__global__ void fill_f32(float* __restrict__ p, float v, int n) {
    const int i = blockIdx.x * blockDim.x + threadIdx.x;
    if (i < n) p[i] = v;
}

// ---------------------------------------------------------------------------
// Host-side orchestration
// ---------------------------------------------------------------------------
static inline int ceil_div(int a, int b) { return (a + b - 1) / b; }

extern "C" void kernel_launch(void* const* d_in, const int* in_sizes, int n_in,
                              void* d_out, int out_size, void* d_ws, size_t ws_size,
                              hipStream_t stream) {
    (void)in_sizes; (void)n_in; (void)out_size; (void)ws_size;

    const float* x     = (const float*)d_in[0];
    const int*   src   = (const int*)  d_in[1];
    const int*   dst   = (const int*)  d_in[2];
    const float* w1_l  = (const float*)d_in[3];
    const float* b1_l  = (const float*)d_in[4];
    const float* w1_r  = (const float*)d_in[5];
    const float* b1_r  = (const float*)d_in[6];
    const float* att1  = (const float*)d_in[7];
    const float* bias1 = (const float*)d_in[8];
    const float* w2_l  = (const float*)d_in[9];
    const float* b2_l  = (const float*)d_in[10];
    const float* w2_r  = (const float*)d_in[11];
    const float* b2_r  = (const float*)d_in[12];
    const float* att2  = (const float*)d_in[13];
    const float* bias2 = (const float*)d_in[14];
    const float* w_lin = (const float*)d_in[15];
    const float* b_lin = (const float*)d_in[16];
    const float* w_fc1 = (const float*)d_in[17];
    const float* b_fc1 = (const float*)d_in[18];
    const float* w_fc2 = (const float*)d_in[19];
    const float* b_fc2 = (const float*)d_in[20];
    float* outp = (float*)d_out;

    // ---- workspace carve-up (64B aligned blocks) ----
    char* base = (char*)d_ws;
    size_t off = 0;
    auto carve = [&](size_t bytes) -> void* {
        off = (off + 63) & ~(size_t)63;
        void* p = base + off;
        off += bytes;
        return p;
    };
    float* xl    = (float*)carve((size_t)NODES * HD * 4);
    float* xr    = (float*)carve((size_t)NODES * HD * 4);
    float* esc   = (float*)carve((size_t)EDGES * H * 4);
    float* emax  = (float*)carve((size_t)NODES * H * 4);
    float* denom = (float*)carve((size_t)NODES * H * 4);
    float* aggr  = (float*)carve((size_t)NODES * HD * 4);
    float* ybuf  = (float*)carve((size_t)NODES * 8 * 4);
    _Float16* xh   = (_Float16*)carve((size_t)NODES * 32 * 2);
    _Float16* h1h  = (_Float16*)carve((size_t)NODES * 96 * 2);   // layer-2 A (K pad 96)
    _Float16* hc   = (_Float16*)carve((size_t)NODES * 160 * 2);  // h1|h2 concat (f16)
    _Float16* yt   = (_Float16*)carve((size_t)BROWS * 64 * 2);   // fc1 A (K pad 64)
    _Float16* z1h  = (_Float16*)carve((size_t)BROWS * 128 * 2);  // fc1 out / fc2 A
    _Float16* bp1l = (_Float16*)carve((size_t)1 * 80 * 32 * 2);
    _Float16* bp1r = (_Float16*)carve((size_t)1 * 80 * 32 * 2);
    _Float16* bp2l = (_Float16*)carve((size_t)3 * 80 * 32 * 2);
    _Float16* bp2r = (_Float16*)carve((size_t)3 * 80 * 32 * 2);
    _Float16* bplin= (_Float16*)carve((size_t)5 * 16 * 32 * 2);
    _Float16* bpfc1= (_Float16*)carve((size_t)2 * 128 * 32 * 2);
    _Float16* bpfc2= (_Float16*)carve((size_t)4 * 48 * 32 * 2);

    const dim3 blk32(32);
    const dim3 blk256(256);
    const int  mt_n    = NODES / 16;   // 4750 (exact)
    const int  mt_b    = BROWS / 16;   // 1000 (exact)
    const int  eh_blk  = ceil_div(EDGES * H, 256);
    const int  nf_blk  = ceil_div(NODES * HD, 256);
    const int  nh_blk  = ceil_div(NODES * H, 256);

    // ---- pack weights & inputs to f16 WMMA layouts ----
    pack_b<<<ceil_div(1 * 80 * 32, 256), blk256, 0, stream>>>(w1_l, bp1l, 8, 80, 1, 80);
    pack_b<<<ceil_div(1 * 80 * 32, 256), blk256, 0, stream>>>(w1_r, bp1r, 8, 80, 1, 80);
    pack_b<<<ceil_div(3 * 80 * 32, 256), blk256, 0, stream>>>(w2_l, bp2l, 80, 80, 3, 80);
    pack_b<<<ceil_div(3 * 80 * 32, 256), blk256, 0, stream>>>(w2_r, bp2r, 80, 80, 3, 80);
    pack_b<<<ceil_div(5 * 16 * 32, 256), blk256, 0, stream>>>(w_lin, bplin, 160, 8, 5, 16);
    pack_b<<<ceil_div(2 * 128 * 32, 256), blk256, 0, stream>>>(w_fc1, bpfc1, 38, 128, 2, 128);
    pack_b<<<ceil_div(4 * 48 * 32, 256), blk256, 0, stream>>>(w_fc2, bpfc2, 128, 37, 4, 48);
    pack_x<<<ceil_div(NODES * 32, 256), blk256, 0, stream>>>(x, xh);

    // ---------------- Layer 1 ----------------
    wmma_gemm_h<1, 0, 0><<<dim3(mt_n, 5), blk32, 0, stream>>>(
        xh, 32, bp1l, 80, b1_l, xl, nullptr, HD, HD);
    wmma_gemm_h<1, 0, 0><<<dim3(mt_n, 5), blk32, 0, stream>>>(
        xh, 32, bp1r, 80, b1_r, xr, nullptr, HD, HD);

    fill_f32<<<nh_blk, blk256, 0, stream>>>(emax, -3.0e38f, NODES * H);
    fill_f32<<<nh_blk, blk256, 0, stream>>>(denom, 0.f, NODES * H);
    fill_f32<<<nf_blk, blk256, 0, stream>>>(aggr, 0.f, NODES * HD);
    fill_f32<<<ceil_div(NODES * 48, 256), blk256, 0, stream>>>((float*)h1h, 0.f, NODES * 48);

    edge_scores<<<eh_blk, blk256, 0, stream>>>(xl, xr, src, dst, att1, esc, emax);
    edge_aggregate<<<eh_blk, blk256, 0, stream>>>(xl, src, dst, esc, emax, denom, aggr);
    node_finalize<<<nf_blk, blk256, 0, stream>>>(aggr, denom, bias1, hc, 160, h1h, 96);

    // ---------------- Layer 2 (A = h1h [N,96] f16) ----------------
    wmma_gemm_h<3, 0, 0><<<dim3(mt_n, 5), blk32, 0, stream>>>(
        h1h, 96, bp2l, 80, b2_l, xl, nullptr, HD, HD);
    wmma_gemm_h<3, 0, 0><<<dim3(mt_n, 5), blk32, 0, stream>>>(
        h1h, 96, bp2r, 80, b2_r, xr, nullptr, HD, HD);

    fill_f32<<<nh_blk, blk256, 0, stream>>>(emax, -3.0e38f, NODES * H);
    fill_f32<<<nh_blk, blk256, 0, stream>>>(denom, 0.f, NODES * H);
    fill_f32<<<nf_blk, blk256, 0, stream>>>(aggr, 0.f, NODES * HD);

    edge_scores<<<eh_blk, blk256, 0, stream>>>(xl, xr, src, dst, att2, esc, emax);
    edge_aggregate<<<eh_blk, blk256, 0, stream>>>(xl, src, dst, esc, emax, denom, aggr);
    node_finalize<<<nf_blk, blk256, 0, stream>>>(aggr, denom, bias2, hc + HD, 160,
                                                 nullptr, 0);

    // ---------------- Head ----------------
    // y = selu(hc @ w_lin + b_lin)   [N,160]@[160,8] -> [N,8] f32
    wmma_gemm_h<5, 1, 0><<<dim3(mt_n, 1), blk32, 0, stream>>>(
        hc, 160, bplin, 16, b_lin, ybuf, nullptr, 8, 8);

    // transpose-gather pack, then z1 = selu(yt @ w_fc1 + b_fc1) -> f16
    pack_yt<<<ceil_div(BROWS * 64, 256), blk256, 0, stream>>>(ybuf, yt);
    wmma_gemm_h<2, 1, 1><<<dim3(mt_b, 8), blk32, 0, stream>>>(
        yt, 64, bpfc1, 128, b_fc1, nullptr, z1h, 128, 128);

    // out = z1 @ w_fc2 + b_fc2   [16000,128]@[128,37] -> f32 d_out
    wmma_gemm_h<4, 0, 0><<<dim3(mt_b, 3), blk32, 0, stream>>>(
        z1h, 128, bpfc2, 48, b_fc2, outp, nullptr, 37, 37);
}
